// InterpretableMultiHeadAttention_21182778704277
// MI455X (gfx1250) — matrix operand order repaired and checked by reading, not compiled, not run on hardware
//
#include <hip/hip_runtime.h>
#include <hip/hip_bf16.h>
#include <math.h>

// ---------------------------------------------------------------------------
// Types for CDNA5 WMMA (wave32): D(16x16 f32) = A(16x32 f16) x B(32x16 f16) + C
// ---------------------------------------------------------------------------
typedef __attribute__((ext_vector_type(16))) _Float16 v16h;
typedef __attribute__((ext_vector_type(4)))  _Float16 v4h;
typedef __attribute__((ext_vector_type(8)))  float    v8f;

#define D_MODEL 1024
#define NHEAD   16
#define HD      64
#define SEQ     2048
#define BATCH   2
#define MTOT    (BATCH * SEQ)          // 4096 rows for the projection GEMMs

// ln(10000) / (HD/2) = 9.210340371976184 / 32
#define ROPE_LN_OVER_HALF 0.28782313662425575f

__device__ __forceinline__ v8f wmma_f16(v16h a, v16h b, v8f c) {
    return __builtin_amdgcn_wmma_f32_16x16x32_f16(
        /*neg_a=*/false, a, /*neg_b=*/false, b,
        /*c_mod=*/(short)0, c, /*reuse_a=*/false, /*reuse_b=*/false);
}

// A fragment: 16x32 f16 tile, row-major source with leading dim ld (elements).
// lanes 0-15: row M=lane, K 0..15; lanes 16-31: K 16..31.
__device__ __forceinline__ v16h load_frag_a(const _Float16* p, int ld, int lane) {
    int row = lane & 15;
    int k0  = (lane >> 4) ? 16 : 0;
    const _Float16* r = p + row * ld;
    v16h a;
#pragma unroll
    for (int i = 0; i < 16; ++i) a[i] = r[k0 + i];
    return a;
}

// B fragment from a B-transposed (N x K row-major) f16 view: Bt[n][k] = B[k][n].
// lane holds column n = lane&15; lanes<16 -> K 0..15, lanes>=16 -> K 16..31.
__device__ __forceinline__ v16h load_frag_bt(const _Float16* p, int ld, int lane) {
    int n  = lane & 15;
    int k0 = (lane >> 4) ? 16 : 0;
    const _Float16* r = p + n * ld + k0;
    v16h b;
#pragma unroll
    for (int i = 0; i < 16; ++i) b[i] = r[i];
    return b;
}

// C/D fragment store: lane holds col n = lane&15; VGPR j -> row j + 8*(lane>=16).
__device__ __forceinline__ void store_frag_c(float* p, int ld, int lane, v8f acc) {
    int n  = lane & 15;
    int mo = (lane >> 4) ? 8 : 0;
#pragma unroll
    for (int j = 0; j < 8; ++j) p[(mo + j) * ld + n] = acc[j];
}

// ---------------------------------------------------------------------------
// Kernel 0: one-shot fp32 -> f16 weight conversion (1024x1024 per matrix)
// grid: 1024 blocks x 256 threads x 4 elements
// ---------------------------------------------------------------------------
__global__ __launch_bounds__(256) void cvt_w_kernel(const float* __restrict__ W,
                                                    _Float16* __restrict__ Wh) {
    int i = (blockIdx.x * 256 + threadIdx.x) * 4;
    float4 f = *(const float4*)(W + i);
    v4h h;
    h[0] = (_Float16)f.x; h[1] = (_Float16)f.y;
    h[2] = (_Float16)f.z; h[3] = (_Float16)f.w;
    *(v4h*)(Wh + i) = h;
}

// ---------------------------------------------------------------------------
// Kernel 1: projection  out = X @ W^T + b, then RoPE (mode 0) or transpose (mode 1)
//   X: [4096][1024] fp32, Wh: [1024][1024] f16 (row-major [N][K] == B^T), b: [1024]
//   mode 0 (Q/K): Out f16 [B][H][T][64], RoPE applied with position t
//   mode 1 (V)  : Out f16 [B][H][64][S]  (transposed for PV B-fragments)
// grid: (MTOT/64, D_MODEL/128), block 256 (8 waves; wave = 1 A-frag x 4 B-frags)
// Double-buffered LDS A tile: one barrier per K-step, staging overlapped w/ WMMA.
// ---------------------------------------------------------------------------
__global__ __launch_bounds__(256) void qkv_proj_rope_kernel(
    const float* __restrict__ X, const _Float16* __restrict__ Wh,
    const float* __restrict__ bias, _Float16* __restrict__ Out, int mode)
{
    __shared__ _Float16 sA[2][64][40];   // double-buffered 64x32 f16 X tile (+pad)
    __shared__ float    sC[64][136];     // 64x128 fp32 output tile (+pad)

    const int tid  = threadIdx.x;
    const int wave = tid >> 5;
    const int lane = tid & 31;
    const int wm   = wave & 3;           // subtile row 0..3
    const int wq   = wave >> 2;          // 0..1 -> columns {wq*4 .. wq*4+3}
    const int mbase = blockIdx.x * 64;
    const int nbase = blockIdx.y * 128;

    v8f acc[4] = {{}, {}, {}, {}};

    // stage first X tile (fp32 -> f16)
    for (int i = tid; i < 64 * 32; i += 256) {
        int r = i >> 5, c = i & 31;
        sA[0][r][c] = (_Float16)X[(size_t)(mbase + r) * D_MODEL + c];
    }
    __syncthreads();

    for (int kk = 0; kk < D_MODEL; kk += 32) {
        int cur = (kk >> 5) & 1;
        // prefetch two steps ahead (gfx1250 global_prefetch_b8)
        if (kk + 64 < D_MODEL)
            __builtin_prefetch(X + (size_t)(mbase + (tid & 63)) * D_MODEL + kk + 64, 0, 0);
        // stage next tile into the other buffer (overlaps with WMMA below)
        if (kk + 32 < D_MODEL) {
            for (int i = tid; i < 64 * 32; i += 256) {
                int r = i >> 5, c = i & 31;
                sA[cur ^ 1][r][c] = (_Float16)X[(size_t)(mbase + r) * D_MODEL + kk + 32 + c];
            }
        }
        v16h a = load_frag_a(&sA[cur][wm * 16][0], 40, lane);
#pragma unroll
        for (int j = 0; j < 4; ++j) {
            v16h b = load_frag_bt(Wh + (size_t)(nbase + (wq * 4 + j) * 16) * D_MODEL + kk,
                                  D_MODEL, lane);
            acc[j] = wmma_f16(a, b, acc[j]);
        }
        __syncthreads();
    }

#pragma unroll
    for (int j = 0; j < 4; ++j)
        store_frag_c(&sC[wm * 16][(wq * 4 + j) * 16], 136, lane, acc[j]);
    __syncthreads();

    if (mode == 0) {
        // bias + RoPE, write [B][H][T][64]
        for (int i = tid; i < 64 * 128; i += 256) {
            int r = i >> 7, c = i & 127;
            int gm = mbase + r;
            int bidx = gm >> 11;          // / SEQ
            int t    = gm & (SEQ - 1);
            int n    = nbase + c;
            int head = n >> 6, d = n & 63;
            float x  = sC[r][c] + bias[n];
            float x2 = sC[r][c ^ 32] + bias[n ^ 32];   // RoPE partner stays in-head
            int   jf = d & 31;
            float f  = (float)t * __expf(-(float)jf * ROPE_LN_OVER_HALF);
            float sv, cv;
            __sincosf(f, &sv, &cv);
            float rot = (d < 32) ? -x2 : x2;
            Out[(((size_t)(bidx * NHEAD + head) * SEQ) + t) * HD + d] =
                (_Float16)(x * cv + rot * sv);
        }
    } else {
        // bias only, write transposed [B][H][64][S]
        for (int i = tid; i < 64 * 128; i += 256) {
            int c = i >> 6, r = i & 63;   // consecutive tid -> consecutive t (coalesced)
            int gm = mbase + r;
            int bidx = gm >> 11;
            int t    = gm & (SEQ - 1);
            int n    = nbase + c;
            int head = n >> 6, d = n & 63;
            float val = sC[r][c] + bias[n];
            Out[(((size_t)(bidx * NHEAD + head) * HD) + d) * SEQ + t] = (_Float16)val;
        }
    }
}

// ---------------------------------------------------------------------------
// Kernel 2: FlashAttention per (b,h). grid: (SEQ/64, B*H), block 256.
//   Q,K: [B][H][S][64] f16 (RoPE'd), Vt: [B][H][64][S] f16
//   Oa : [4096][1024] f16 attention output (laid out for final projection)
// q is pre-scaled by 1/8 (= Hd^-0.5, exact power of two in f16).
// ---------------------------------------------------------------------------
__global__ __launch_bounds__(256) void flash_attn_kernel(
    const _Float16* __restrict__ Q, const _Float16* __restrict__ K,
    const _Float16* __restrict__ Vt, _Float16* __restrict__ Oa)
{
    __shared__ _Float16 sQ[64][72];
    __shared__ float    sS[64][68];
    __shared__ _Float16 sP[64][72];
    __shared__ float    sM[64], sL[64], sAl[64];

    const int tid  = threadIdx.x;
    const int wave = tid >> 5;
    const int lane = tid & 31;
    const int wm   = wave & 3;
    const int wn   = wave >> 2;
    const int bh   = blockIdx.y;             // 0..31
    const int b    = bh >> 4, h = bh & 15;
    const int t0   = blockIdx.x * 64;

    const _Float16* q  = Q  + (size_t)bh * SEQ * HD;
    const _Float16* k  = K  + (size_t)bh * SEQ * HD;
    const _Float16* vt = Vt + (size_t)bh * HD * SEQ;

    // stage q tile, folding in softmax scale 1/sqrt(64) = 0.125
    for (int i = tid; i < 64 * 64; i += 256) {
        int r = i >> 6, c = i & 63;
        sQ[r][c] = (_Float16)((float)q[(size_t)(t0 + r) * HD + c] * 0.125f);
    }
    if (tid < 64) { sM[tid] = -INFINITY; sL[tid] = 0.f; }
    __syncthreads();

    v8f o0 = {}, o1 = {};

    for (int s0 = 0; s0 < SEQ; s0 += 64) {
        // ---- scores: S = qs @ k^T  (M=64,N=64,K=Hd=64) ----
        v8f sc0 = {}, sc1 = {};
#pragma unroll
        for (int kk = 0; kk < HD; kk += 32) {
            v16h a  = load_frag_a(&sQ[wm * 16][kk], 72, lane);
            v16h b0 = load_frag_bt(k + (size_t)(s0 + (wn * 2 + 0) * 16) * HD + kk, HD, lane);
            v16h b1 = load_frag_bt(k + (size_t)(s0 + (wn * 2 + 1) * 16) * HD + kk, HD, lane);
            sc0 = wmma_f16(a, b0, sc0);
            sc1 = wmma_f16(a, b1, sc1);
        }
        store_frag_c(&sS[wm * 16][(wn * 2 + 0) * 16], 68, lane, sc0);
        store_frag_c(&sS[wm * 16][(wn * 2 + 1) * 16], 68, lane, sc1);
        __syncthreads();

        // ---- online softmax: 4 threads per row (within one wave) ----
        {
            int row = tid >> 2, part = tid & 3;
            float mx = -INFINITY;
#pragma unroll
            for (int c = 0; c < 16; ++c) mx = fmaxf(mx, sS[row][part * 16 + c]);
            mx = fmaxf(mx, __shfl_xor(mx, 1, 4));
            mx = fmaxf(mx, __shfl_xor(mx, 2, 4));
            float m_old = sM[row];
            float m_new = fmaxf(m_old, mx);
            float sum = 0.f;
#pragma unroll
            for (int c = 0; c < 16; ++c) {
                float p = __expf(sS[row][part * 16 + c] - m_new);
                sP[row][part * 16 + c] = (_Float16)p;
                sum += p;
            }
            sum += __shfl_xor(sum, 1, 4);
            sum += __shfl_xor(sum, 2, 4);
            if (part == 0) {
                float alpha = __expf(m_old - m_new);   // 0 on first tile (m_old = -inf)
                sAl[row] = alpha;
                sL[row]  = sL[row] * alpha + sum;
                sM[row]  = m_new;
            }
        }
        __syncthreads();

        // ---- rescale O accumulators, then O += P @ V_tile ----
        {
            int mo = (lane >> 4) ? 8 : 0;
#pragma unroll
            for (int j = 0; j < 8; ++j) {
                float al = sAl[wm * 16 + mo + j];
                o0[j] *= al;
                o1[j] *= al;
            }
        }
#pragma unroll
        for (int kk = 0; kk < 64; kk += 32) {
            v16h a  = load_frag_a(&sP[wm * 16][kk], 72, lane);
            v16h b0 = load_frag_bt(vt + (size_t)((wn * 2 + 0) * 16) * SEQ + s0 + kk, SEQ, lane);
            v16h b1 = load_frag_bt(vt + (size_t)((wn * 2 + 1) * 16) * SEQ + s0 + kk, SEQ, lane);
            o0 = wmma_f16(a, b0, o0);
            o1 = wmma_f16(a, b1, o1);
        }
        __syncthreads();
    }

    // ---- epilogue: divide by l, write f16 [4096][1024] for output projection ----
    store_frag_c(&sS[wm * 16][(wn * 2 + 0) * 16], 68, lane, o0);
    store_frag_c(&sS[wm * 16][(wn * 2 + 1) * 16], 68, lane, o1);
    __syncthreads();
    for (int i = tid; i < 64 * 64; i += 256) {
        int r = i >> 6, d = i & 63;
        float val = sS[r][d] / sL[r];
        size_t gm = (size_t)b * SEQ + t0 + r;
        Oa[gm * D_MODEL + h * HD + d] = (_Float16)val;
    }
}

// ---------------------------------------------------------------------------
// Kernel 3: final projection  out = A @ Wo^T + bo  (fp32 output)
//   A: [4096][1024] f16, Who: [1024][1024] f16, Out: [4096][1024] fp32
// grid: (MTOT/64, D_MODEL/128), block 256; same double-buffered scheme
// ---------------------------------------------------------------------------
__global__ __launch_bounds__(256) void out_proj_kernel(
    const _Float16* __restrict__ A, const _Float16* __restrict__ Who,
    const float* __restrict__ bo, float* __restrict__ Out)
{
    __shared__ _Float16 sA[2][64][40];

    const int tid  = threadIdx.x;
    const int wave = tid >> 5;
    const int lane = tid & 31;
    const int wm   = wave & 3;
    const int wq   = wave >> 2;
    const int mbase = blockIdx.x * 64;
    const int nbase = blockIdx.y * 128;

    v8f acc[4] = {{}, {}, {}, {}};

    for (int i = tid; i < 64 * 32; i += 256) {
        int r = i >> 5, c = i & 31;
        sA[0][r][c] = A[(size_t)(mbase + r) * D_MODEL + c];
    }
    __syncthreads();

    for (int kk = 0; kk < D_MODEL; kk += 32) {
        int cur = (kk >> 5) & 1;
        if (kk + 64 < D_MODEL)
            __builtin_prefetch(A + (size_t)(mbase + (tid & 63)) * D_MODEL + kk + 64, 0, 0);
        if (kk + 32 < D_MODEL) {
            for (int i = tid; i < 64 * 32; i += 256) {
                int r = i >> 5, c = i & 31;
                sA[cur ^ 1][r][c] = A[(size_t)(mbase + r) * D_MODEL + kk + 32 + c];
            }
        }
        v16h a = load_frag_a(&sA[cur][wm * 16][0], 40, lane);
#pragma unroll
        for (int j = 0; j < 4; ++j) {
            v16h b = load_frag_bt(Who + (size_t)(nbase + (wq * 4 + j) * 16) * D_MODEL + kk,
                                  D_MODEL, lane);
            acc[j] = wmma_f16(a, b, acc[j]);
        }
        __syncthreads();
    }

    // direct coalesced fp32 stores + bias
    int mo = (lane >> 4) ? 8 : 0;
    int mrow = mbase + wm * 16 + mo;
#pragma unroll
    for (int j = 0; j < 4; ++j) {
        int n = nbase + (wq * 4 + j) * 16 + (lane & 15);
        float bb = bo[n];
#pragma unroll
        for (int r = 0; r < 8; ++r)
            Out[(size_t)(mrow + r) * D_MODEL + n] = acc[j][r] + bb;
    }
}

// ---------------------------------------------------------------------------
// Host-side launcher
// ---------------------------------------------------------------------------
extern "C" void kernel_launch(void* const* d_in, const int* in_sizes, int n_in,
                              void* d_out, int out_size, void* d_ws, size_t ws_size,
                              hipStream_t stream) {
    const float* query = (const float*)d_in[0];
    const float* key_  = (const float*)d_in[1];
    const float* value = (const float*)d_in[2];
    const float* Wq    = (const float*)d_in[3];
    const float* bq    = (const float*)d_in[4];
    const float* Wk    = (const float*)d_in[5];
    const float* bk    = (const float*)d_in[6];
    const float* Wv    = (const float*)d_in[7];
    const float* bv    = (const float*)d_in[8];
    const float* Wo    = (const float*)d_in[9];
    const float* bo    = (const float*)d_in[10];

    const size_t elems = (size_t)MTOT * D_MODEL;      // 4096*1024 halves per tensor
    const size_t welem = (size_t)D_MODEL * D_MODEL;   // 1024*1024 halves per weight
    _Float16* qb  = (_Float16*)d_ws;                  // [B][H][T][64]  (RoPE'd)
    _Float16* kb  = qb  + elems;                      // [B][H][S][64]  (RoPE'd)
    _Float16* vtb = kb  + elems;                      // [B][H][64][S]  (transposed V)
    _Float16* ab  = vtb + elems;                      // [4096][1024]   attention output
    _Float16* whq = ab  + elems;                      // f16 weights
    _Float16* whk = whq + welem;
    _Float16* whv = whk + welem;
    _Float16* who = whv + welem;

    dim3 blk(256);

    dim3 gCvt(welem / (256 * 4));                     // 1024 blocks
    cvt_w_kernel<<<gCvt, blk, 0, stream>>>(Wq, whq);
    cvt_w_kernel<<<gCvt, blk, 0, stream>>>(Wk, whk);
    cvt_w_kernel<<<gCvt, blk, 0, stream>>>(Wv, whv);
    cvt_w_kernel<<<gCvt, blk, 0, stream>>>(Wo, who);

    dim3 gProj(MTOT / 64, D_MODEL / 128);
    qkv_proj_rope_kernel<<<gProj, blk, 0, stream>>>(query, whq, bq, qb, 0);
    qkv_proj_rope_kernel<<<gProj, blk, 0, stream>>>(key_,  whk, bk, kb, 0);
    qkv_proj_rope_kernel<<<gProj, blk, 0, stream>>>(value, whv, bv, vtb, 1);

    dim3 gAttn(SEQ / 64, BATCH * NHEAD);
    flash_attn_kernel<<<gAttn, blk, 0, stream>>>(qb, kb, vtb, ab);

    dim3 gOut(MTOT / 64, D_MODEL / 128);
    out_proj_kernel<<<gOut, blk, 0, stream>>>(ab, who, bo, (float*)d_out);
}
